// AutoCorrelation_loss_V_19877108646108
// MI455X (gfx1250) — matrix operand order repaired and checked by reading, not compiled, not run on hardware
//
#include <hip/hip_runtime.h>
#include <cstdint>
#include <cstddef>

// ======================= CDNA5 types =======================
typedef __attribute__((ext_vector_type(16))) __bf16 v16bf;
typedef __attribute__((ext_vector_type(8)))  float  v8f;
typedef __attribute__((ext_vector_type(4)))  unsigned int v4u;
typedef __attribute__((ext_vector_type(8)))  int    v8i;
typedef __attribute__((ext_vector_type(4)))  int    v4i;
typedef __attribute__((ext_vector_type(4)))  float  v4f;
typedef __attribute__((ext_vector_type(4)))  unsigned short v4us;

#if defined(__HIP_DEVICE_COMPILE__) && __has_builtin(__builtin_amdgcn_tensor_load_to_lds) && __has_builtin(__builtin_amdgcn_s_wait_tensorcnt)
#define HAVE_TDM 1
#else
#define HAVE_TDM 0
#endif

// ======================= problem constants =======================
// queries/keys/values: (B=4, L=2048, H=8, E=64) f32.  C = H*E = 512, N = 2L = 4096.
#define PB 4
#define PL 2048
#define PC 512
#define PN 4096
#define PH 8
#define PE 64
#define TOPK 7
#define NCHUNK 16
#define CHUNKL 128
#define ROWB 1024            // bytes per z row (512 bf16)
#define LDSROWB 1040         // padded LDS row stride (16B pad kills bank conflicts)
#define NJT 256              // 4096/16 column tiles

// ws layout (bytes)
#define OFF_ZB   0ull                     // bf16 z, B x 4096 x 512  (16 MiB)
#define OFF_LSE  16777216ull              // f32 [4][4096]
#define OFF_POS  16842752ull              // f32 [4][2048]
#define OFF_INST 16875520ull              // f32 [4][2048]
#define OFF_CSUM 16908288ull              // f32 [4*8][16][64]
#define OFF_COFF 17039360ull              // f32 [4*8][16][64]
#define OFF_IDX  17170432ull              // int [7]

__device__ __forceinline__ unsigned short f2bf(float f) {
  unsigned u = __float_as_uint(f);
  unsigned r = (u + 0x7FFFu + ((u >> 16) & 1u)) >> 16;  // RNE
  return (unsigned short)r;
}
__device__ __forceinline__ v8f vz8() {
  v8f r;
  #pragma unroll
  for (int i = 0; i < 8; ++i) r[i] = 0.f;
  return r;
}

// ======================= K1: pack q,k -> bf16 z =======================
__global__ __launch_bounds__(256) void k_pack(const float* __restrict__ q,
                                              const float* __restrict__ k,
                                              unsigned short* __restrict__ zb) {
  size_t i = (size_t)blockIdx.x * 256 + threadIdx.x;      // 0 .. 2*1048576-1 (vec4 units)
  bool isK = i >= 1048576u;
  size_t vi = isK ? i - 1048576u : i;
  const v4f* src = (const v4f*)(isK ? k : q);
  v4f v = src[vi];
  size_t e = vi * 4;
  size_t b = e >> 20;                                     // 2048*512 = 2^20 per batch
  size_t dstE = e + ((b + (isK ? 1 : 0)) << 20);          // per batch: 2048 q rows then 2048 k rows
  v4us o; o[0] = f2bf(v[0]); o[1] = f2bf(v[1]); o[2] = f2bf(v[2]); o[3] = f2bf(v[3]);
  *(v4us*)(zb + dstE) = o;
}

// ======================= K2: temporal LSE via WMMA + TDM =======================
__device__ __forceinline__ void stage_tile(const unsigned char* zbB, unsigned char* dstLds,
                                           int jt, int tid) {
  const unsigned char* src = zbB + (size_t)jt * (16 * ROWB);
#if HAVE_TDM
  if (tid < 32) {  // one wave issues the TDM op (EXEC ignored by tensor ops)
    unsigned long long ga = (unsigned long long)src;
    unsigned loff = (unsigned)(unsigned long long)(void*)dstLds; // low 32 bits = LDS offset
    v4u g0; v8i g1; v4i g2, g3; v8i g4;
    // D# group0: count=1 | lds_addr | global_addr(57b) | type=2
    g0[0] = 1u;
    g0[1] = loff;
    g0[2] = (unsigned)ga;
    g0[3] = (unsigned)((ga >> 32) & 0x1FFFFFFull) | 0x80000000u;
    // D# group1: data_size=1B, pad_enable, pad_interval=7 (every 1024B), pad_amount=3 (16B)
    g1[0] = (int)((1u << 20) | (7u << 22) | (3u << 25));
    g1[1] = (int)(1024u << 16);   // tensor_dim0[15:0] = 1024 bytes
    g1[2] = (int)(4096u << 16);   // tensor_dim1[15:0] = 4096 rows
    g1[3] = (int)(1024u << 16);   // tile_dim0 = 1024 bytes
    g1[4] = 16;                   // tile_dim1 = 16 rows
    g1[5] = 1024;                 // tensor_dim0_stride = 1024 bytes
    g1[6] = 0; g1[7] = 0;
    g2[0] = g2[1] = g2[2] = g2[3] = 0; g3 = g2;
    #pragma unroll
    for (int i = 0; i < 8; ++i) g4[i] = 0;   // 2D tile: groups 2/3 unused
    __builtin_amdgcn_tensor_load_to_lds(g0, g1, g2, g3, g4, 0);
  }
#else
  // cooperative copy fallback: 16KB as 1024 x 16B chunks, with LDS row padding
  for (int qd = tid; qd < 1024; qd += 256) {
    int row = qd >> 6, inrow = (qd & 63) << 4;
    *(v4u*)(dstLds + row * LDSROWB + inrow) = *(const v4u*)(src + ((size_t)qd << 4));
  }
#endif
}

union ABu { v16bf v; v4u q[2]; };

__global__ __launch_bounds__(256) void k_lse(const unsigned char* __restrict__ zb,
                                             float* __restrict__ lse) {
  __shared__ __align__(16) unsigned char Bbuf[2][16 * LDSROWB];
  const int tid = threadIdx.x, lane = tid & 31, wid = tid >> 5;
  const int b = blockIdx.x >> 5;              // 4 batches x 32 panels
  const int panel = blockIdx.x & 31;
  const int rowbase = panel * 128 + wid * 16; // 16 rows per wave
  const int l15 = lane & 15, hs = lane >> 4;

  const unsigned char* zbB = zb + (size_t)b * ((size_t)PN * ROWB);

  // A fragments resident in VGPRs: 16 rows x K=512 bf16 (16 frags of 16x32)
  // A layout: lanes 0-15 row m, K 0..15; lanes 16-31 row m, K 16..31
  v16bf a[16];
  {
    const unsigned char* arow = zbB + (size_t)(rowbase + l15) * ROWB + hs * 32;
    #pragma unroll
    for (int kt = 0; kt < 16; ++kt) {
      ABu u;
      u.q[0] = *(const v4u*)(arow + kt * 64);
      u.q[1] = *(const v4u*)(arow + kt * 64 + 16);
      a[kt] = u.v;
    }
  }

  // per-lane online logsumexp state: reg r holds rows rowbase+r (lanes<16) / rowbase+r+8
  float m[8], s[8];
  #pragma unroll
  for (int r = 0; r < 8; ++r) { m[r] = -1e30f; s[r] = 0.f; }

  stage_tile(zbB, &Bbuf[0][0], 0, tid);
  for (int jt = 0; jt < NJT; ++jt) {
#if HAVE_TDM
    if (tid < 32) __builtin_amdgcn_s_wait_tensorcnt(0);
#endif
    __syncthreads();                           // tile jt ready; tile jt-1 fully consumed
    if (jt + 1 < NJT) stage_tile(zbB, &Bbuf[(jt + 1) & 1][0], jt + 1, tid);

    // B layout: lane n = column n (z row colbase+n); lanes 0-15 K 0..15, lanes 16-31 K 16..31
    const unsigned char* bb = &Bbuf[jt & 1][0] + l15 * LDSROWB + hs * 32;
    v8f acc0 = vz8(), acc1 = vz8();

    // software-pipelined K loop: two B fragments in flight ahead of the WMMAs
    ABu u0, u1;
    u0.q[0] = *(const v4u*)(bb + 0 * 64);
    u0.q[1] = *(const v4u*)(bb + 0 * 64 + 16);
    u1.q[0] = *(const v4u*)(bb + 1 * 64);
    u1.q[1] = *(const v4u*)(bb + 1 * 64 + 16);
    #pragma unroll
    for (int kt = 0; kt < 16; kt += 2) {
      ABu n0, n1;
      if (kt + 2 < 16) {
        n0.q[0] = *(const v4u*)(bb + (kt + 2) * 64);
        n0.q[1] = *(const v4u*)(bb + (kt + 2) * 64 + 16);
        n1.q[0] = *(const v4u*)(bb + (kt + 3) * 64);
        n1.q[1] = *(const v4u*)(bb + (kt + 3) * 64 + 16);
      }
      acc0 = __builtin_amdgcn_wmma_f32_16x16x32_bf16(false, a[kt],     false, u0.v, (short)0, acc0, false, false);
      acc1 = __builtin_amdgcn_wmma_f32_16x16x32_bf16(false, a[kt + 1], false, u1.v, (short)0, acc1, false, false);
      u0 = n0; u1 = n1;
    }
    v8f d = acc0 + acc1;

    const int c0 = jt * 16;
    const bool diag = (c0 == rowbase);         // 16-aligned tiles: diagonal only when equal
    #pragma unroll
    for (int r = 0; r < 8; ++r) {
      float x = d[r];
      if (diag && (lane == r || lane == r + 24)) x = -1e30f;  // exclude sim[i,i]
      float nm = fmaxf(m[r], x);
      s[r] = s[r] * __expf(m[r] - nm) + __expf(x - nm);
      m[r] = nm;
    }
  }

  // merge (m,s) across the 16 lanes of each half-wave, then write LSE
  #pragma unroll
  for (int r = 0; r < 8; ++r) {
    #pragma unroll
    for (int off = 1; off < 16; off <<= 1) {
      float mo = __shfl_xor(m[r], off, 32);
      float so = __shfl_xor(s[r], off, 32);
      float nm = fmaxf(m[r], mo);
      s[r] = s[r] * __expf(m[r] - nm) + so * __expf(mo - nm);
      m[r] = nm;
    }
    if (l15 == 0)
      lse[(size_t)b * PN + rowbase + r + hs * 8] = m[r] + __logf(s[r]);
  }
}

// ======================= K3: temporal positives (fp32) =======================
__global__ __launch_bounds__(256) void k_pos(const float* __restrict__ q,
                                             const float* __restrict__ k,
                                             float* __restrict__ pos) {
  int lane = threadIdx.x & 31, wid = threadIdx.x >> 5;
  int gw = blockIdx.x * 8 + wid;            // 0..8191
  int b = gw >> 11, t = gw & (PL - 1);
  const float* qr = q + ((size_t)(b * PL + t)) * PC;
  const float* kr = k + ((size_t)(b * PL + t)) * PC;
  float s = 0.f;
  #pragma unroll
  for (int u = 0; u < 16; ++u) s = fmaf(qr[lane + 32 * u], kr[lane + 32 * u], s);
  #pragma unroll
  for (int off = 16; off >= 1; off >>= 1) s += __shfl_xor(s, off, 32);
  if (lane == 0) pos[b * PL + t] = s;
}

// ======================= K4: instance loss (8x8 Gram per t, fp32) =======================
__device__ __forceinline__ constexpr int triIdx(int i, int j) { return i * 7 - i * (i - 1) / 2 + (j - i - 1); }
__device__ __forceinline__ float simAt(const float* acc, int i, int j) {
  return (i < j) ? acc[triIdx(i, j)] : acc[triIdx(j, i)];
}
__device__ __forceinline__ float rowLSE(const float* acc, int r) {
  float m = -3e38f;
  #pragma unroll
  for (int j = 0; j < 8; ++j) if (j != r) m = fmaxf(m, simAt(acc, r, j));
  float ss = 0.f;
  #pragma unroll
  for (int j = 0; j < 8; ++j) if (j != r) ss += __expf(simAt(acc, r, j) - m);
  return m + __logf(ss);
}

__global__ __launch_bounds__(256) void k_inst(const float* __restrict__ q,
                                              const float* __restrict__ k,
                                              float* __restrict__ inst) {
  int lane = threadIdx.x & 31, wid = threadIdx.x >> 5;
  int t = blockIdx.x * 8 + wid;             // 0..2047
  float acc[28];
  #pragma unroll
  for (int p = 0; p < 28; ++p) acc[p] = 0.f;
  #pragma unroll
  for (int u = 0; u < 16; ++u) {
    int c = lane + 32 * u;
    float zv[8];
    #pragma unroll
    for (int i = 0; i < 8; ++i) {
      const float* src = (i < 4) ? (q + ((size_t)(i * PL + t)) * PC)
                                 : (k + ((size_t)((i - 4) * PL + t)) * PC);
      zv[i] = src[c];
    }
    #pragma unroll
    for (int i = 0; i < 8; ++i) {
      #pragma unroll
      for (int j = i + 1; j < 8; ++j)
        acc[triIdx(i, j)] = fmaf(zv[i], zv[j], acc[triIdx(i, j)]);
    }
  }
  #pragma unroll
  for (int p = 0; p < 28; ++p) {
    #pragma unroll
    for (int off = 16; off >= 1; off >>= 1) acc[p] += __shfl_xor(acc[p], off, 32);
  }
  if (lane == 0) {
    #pragma unroll
    for (int b = 0; b < 4; ++b) {
      float l1 = rowLSE(acc, b), l2 = rowLSE(acc, 4 + b);
      float ps = simAt(acc, b, 4 + b);
      inst[b * PL + t] = 0.5f * (l1 + l2) - ps;
    }
  }
}

// ======================= K5: combine + top-7 (desc, lower index wins ties) ============
__global__ __launch_bounds__(256) void k_topk(const float* __restrict__ lse,
                                              const float* __restrict__ pos,
                                              const float* __restrict__ inst,
                                              int* __restrict__ idxout) {
  __shared__ float sc[PL];
  __shared__ float bv[256];
  __shared__ int   bi[256];
  int tid = threadIdx.x;
  for (int t = tid; t < PL; t += 256) {
    float acc = 0.f;
    #pragma unroll
    for (int b = 0; b < 4; ++b) {
      float temp = 0.5f * (lse[b * PN + t] + lse[b * PN + PL + t]) - pos[b * PL + t];
      acc += 0.5f * inst[b * PL + t] + 0.5f * temp;
    }
    sc[t] = 0.25f * acc;
  }
  __syncthreads();
  for (int kk = 0; kk < TOPK; ++kk) {
    float best = -3e38f; int bidx = 0x7fffffff;
    for (int t = tid; t < PL; t += 256) {
      float v = sc[t];
      if (v > best || (v == best && t < bidx)) { best = v; bidx = t; }
    }
    bv[tid] = best; bi[tid] = bidx;
    __syncthreads();
    for (int sft = 128; sft > 0; sft >>= 1) {
      if (tid < sft) {
        float vo = bv[tid + sft]; int io = bi[tid + sft];
        if (vo > bv[tid] || (vo == bv[tid] && io < bi[tid])) { bv[tid] = vo; bi[tid] = io; }
      }
      __syncthreads();
    }
    if (tid == 0) { idxout[kk] = bi[0]; sc[bi[0]] = -3e38f; }
    __syncthreads();
  }
}

// ======================= K6/K7/K8: chunked cumsum of V with scaled rows ===============
__global__ __launch_bounds__(64) void k_csum(const float* __restrict__ V, float* __restrict__ csum) {
  int id = blockIdx.x;                 // (b*H+h)*NCHUNK + c
  int c = id & (NCHUNK - 1), bh = id >> 4;
  int b = bh >> 3, h = bh & 7, e = threadIdx.x;
  const float* vp = V + (((size_t)b * PL + (size_t)c * CHUNKL) * PH + h) * PE + e;
  float s = 0.f;
  for (int l = 0; l < CHUNKL; ++l) s += vp[(size_t)l * (PH * PE)];
  csum[((size_t)bh * NCHUNK + c) * PE + e] = s;
}

__global__ __launch_bounds__(64) void k_coff(const float* __restrict__ csum, float* __restrict__ coff) {
  int id = blockIdx.x * 64 + threadIdx.x;  // 0..2047 : (bh, e)
  int bh = id >> 6, e = id & 63;
  float run = 0.f;
  for (int c = 0; c < NCHUNK; ++c) {
    coff[((size_t)bh * NCHUNK + c) * PE + e] = run;
    run += csum[((size_t)bh * NCHUNK + c) * PE + e];
  }
}

__global__ __launch_bounds__(64) void k_out(const float* __restrict__ V,
                                            const float* __restrict__ coff,
                                            const int* __restrict__ idx,
                                            float* __restrict__ out) {
  int id = blockIdx.x;
  int c = id & (NCHUNK - 1), bh = id >> 4;
  int b = bh >> 3, h = bh & 7, e = threadIdx.x;
  int i0 = idx[0], i1 = idx[1], i2 = idx[2], i3 = idx[3], i4 = idx[4], i5 = idx[5], i6 = idx[6];
  const float* vp = V + (((size_t)b * PL + (size_t)c * CHUNKL) * PH + h) * PE + e;
  float* op = out + ((size_t)bh * PL + (size_t)c * CHUNKL) * PE + e;
  float run = coff[((size_t)bh * NCHUNK + c) * PE + e];
  for (int l0 = 0; l0 < CHUNKL; ++l0) {
    int l = c * CHUNKL + l0;
    run += vp[(size_t)l0 * (PH * PE)];
    float o = run;
    bool hit = (l == i0) | (l == i1) | (l == i2) | (l == i3) | (l == i4) | (l == i5) | (l == i6);
    if (hit) o = run / (float)(l + 1);     // uniform softmax over l<=index -> cumsum/(idx+1)
    op[(size_t)l0 * PE] = o;
  }
}

// ======================= host launcher =======================
extern "C" void kernel_launch(void* const* d_in, const int* in_sizes, int n_in,
                              void* d_out, int out_size, void* d_ws, size_t ws_size,
                              hipStream_t stream) {
  (void)in_sizes; (void)n_in; (void)out_size; (void)ws_size;
  const float* q = (const float*)d_in[0];
  const float* k = (const float*)d_in[1];
  const float* v = (const float*)d_in[2];
  char* ws = (char*)d_ws;
  unsigned short* zb = (unsigned short*)(ws + OFF_ZB);
  float* lse  = (float*)(ws + OFF_LSE);
  float* pos  = (float*)(ws + OFF_POS);
  float* inst = (float*)(ws + OFF_INST);
  float* csum = (float*)(ws + OFF_CSUM);
  float* coff = (float*)(ws + OFF_COFF);
  int*   idx  = (int*)(ws + OFF_IDX);
  float* out  = (float*)d_out;

  k_pack<<<8192, 256, 0, stream>>>(q, k, zb);
  k_lse <<<128,  256, 0, stream>>>((const unsigned char*)zb, lse);
  k_pos <<<1024, 256, 0, stream>>>(q, k, pos);
  k_inst<<<256,  256, 0, stream>>>(q, k, inst);
  k_topk<<<1,    256, 0, stream>>>(lse, pos, inst, idx);
  k_csum<<<512,  64,  0, stream>>>(v, csum);
  k_coff<<<32,   64,  0, stream>>>(csum, coff);
  k_out <<<512,  64,  0, stream>>>(v, coff, idx, out);
}
